// TransformerBlock_91070486544811
// MI455X (gfx1250) — compile-verified, matrix operand order checked by reading, not run on hardware
//
#include <hip/hip_runtime.h>
#include <hip/hip_bf16.h>

// ---------------------------------------------------------------------------
// Types for CDNA5 WMMA
// ---------------------------------------------------------------------------
typedef __attribute__((ext_vector_type(16))) __bf16 bf16x16;
typedef __attribute__((ext_vector_type(8)))  float  f32x8;

union Frag {
    bf16x16      v;
    unsigned int u[8];
};

__device__ __forceinline__ unsigned short f2bf(float f) {
    unsigned int u = __float_as_uint(f);
    u += 0x7FFFu + ((u >> 16) & 1u);     // round-to-nearest-even
    return (unsigned short)(u >> 16);
}

// uint-column inside a 16-uint (32 bf16) k-slab for the 16-bit A/B fragment
// layout (ISA 7.12.2): VGPR j holds K={2j,2j+1}+8*half for j<4, K=16+2(j-4)+8*half for j>=4
__device__ __forceinline__ int kcol(int j, int hf) {
    return (j & 3) + ((j >> 2) << 3) + (hf << 2);
}

#define WMMA_BF16(A, B, C) \
    __builtin_amdgcn_wmma_f32_16x16x32_bf16(false, (A), false, (B), (short)0, (C), false, false)

// Async global -> LDS copy (CDNA5, tracked by ASYNCcnt).
// vdst = LDS byte offset (low 32 bits of the flat shared address per ISA 10.2).
__device__ __forceinline__ void async_b128(unsigned lds_off, const void* gaddr) {
    asm volatile("global_load_async_to_lds_b128 %0, %1, off"
                 :: "v"(lds_off), "v"(gaddr) : "memory");
}

// ---------------------------------------------------------------------------
// fp32 -> bf16 conversion (weights)
// ---------------------------------------------------------------------------
__global__ void cvt_f32_bf16(const float* __restrict__ in, unsigned short* __restrict__ out, int n) {
    int i = blockIdx.x * blockDim.x + threadIdx.x;
    int stride = gridDim.x * blockDim.x;
    for (; i < n; i += stride) out[i] = f2bf(in[i]);
}

// ---------------------------------------------------------------------------
// LayerNorm (row = 1024), fp32 in -> bf16 out
// ---------------------------------------------------------------------------
__global__ __launch_bounds__(256) void ln_bf16(const float* __restrict__ x,
                                               const float* __restrict__ g,
                                               const float* __restrict__ bt,
                                               unsigned short* __restrict__ out) {
    __shared__ float rs[256];
    __shared__ float rq[256];
    const int row = blockIdx.x;
    const int t   = threadIdx.x;
    float v[4];
    float s = 0.f, q = 0.f;
#pragma unroll
    for (int i = 0; i < 4; ++i) {
        v[i] = x[(size_t)row * 1024 + t + i * 256];
        s += v[i];
        q += v[i] * v[i];
    }
    rs[t] = s; rq[t] = q;
    __syncthreads();
    for (int o = 128; o > 0; o >>= 1) {
        if (t < o) { rs[t] += rs[t + o]; rq[t] += rq[t + o]; }
        __syncthreads();
    }
    const float mu   = rs[0] * (1.0f / 1024.0f);
    const float var  = rq[0] * (1.0f / 1024.0f) - mu * mu;
    const float rstd = rsqrtf(var + 1e-5f);
#pragma unroll
    for (int i = 0; i < 4; ++i) {
        int c = t + i * 256;
        out[(size_t)row * 1024 + c] = f2bf((v[i] - mu) * rstd * g[c] + bt[c]);
    }
}

// ---------------------------------------------------------------------------
// Tiled bf16 WMMA GEMM:  C[M,N] = A[M,K] @ W[N,K]^T (+bias)(+gelu)(+resid)
// block = 256 threads (8 waves) -> 256x128 tile; wave tile = 64x64; k-step 32.
// Double-buffered LDS filled by async global->LDS (ASYNCcnt) copies.
// Epilogue specialized at compile time to keep it branch-free.
// ---------------------------------------------------------------------------
template <bool HAS_BIAS, bool ACT_GELU, bool HAS_RESID, bool OUT_F, bool OUT_B>
__global__ __launch_bounds__(256) void gemm_bf16(const unsigned short* __restrict__ A,
                                                 const unsigned short* __restrict__ W,
                                                 const float* __restrict__ bias,
                                                 const float* __restrict__ resid,
                                                 float* __restrict__ outF,
                                                 unsigned short* __restrict__ outB,
                                                 int M, int N, int K) {
    __shared__ __attribute__((aligned(16))) unsigned int lA[2][256 * 16];  // 2 x 16KB
    __shared__ __attribute__((aligned(16))) unsigned int lB[2][128 * 16];  // 2 x 8KB
    const unsigned int* A_u = (const unsigned int*)A;
    const unsigned int* W_u = (const unsigned int*)W;

    const int tid  = threadIdx.x;
    const int wave = tid >> 5;
    const int lane = tid & 31;
    const int hf   = lane >> 4;
    const int ml   = lane & 15;
    const int wm   = wave >> 1;   // 0..3 -> 64-row strip
    const int wn   = wave & 1;    // 0..1 -> 64-col strip
    const int m0   = blockIdx.y * 256;
    const int n0   = blockIdx.x * 128;
    const int Ku   = K >> 1;

    f32x8 acc[4][4] = {};

    // issue async loads for one 32-wide k-slab into buffer `buf`
    auto issue = [&](int buf, int ku0) {
#pragma unroll
        for (int i = 0; i < 4; ++i) {                       // A: 1024 b128 slots
            int slot = i * 256 + tid;
            int r = slot >> 2, c4 = (slot & 3) << 2;
            unsigned lofs = (unsigned)(size_t)&lA[buf][slot * 4];
            async_b128(lofs, &A_u[(size_t)(m0 + r) * Ku + ku0 + c4]);
        }
#pragma unroll
        for (int i = 0; i < 2; ++i) {                       // B: 512 b128 slots
            int slot = i * 256 + tid;
            int r = slot >> 2, c4 = (slot & 3) << 2;
            unsigned lofs = (unsigned)(size_t)&lB[buf][slot * 4];
            async_b128(lofs, &W_u[(size_t)(n0 + r) * Ku + ku0 + c4]);
        }
    };

    const int nsteps = K >> 5;
    issue(0, 0);

    for (int s = 0; s < nsteps; ++s) {
        const int cur = s & 1;
        __syncthreads();                       // everyone done reading buf[cur^1]
        if (s + 1 < nsteps) {
            issue(cur ^ 1, (s + 1) << 4);      // prefetch next slab (16 uints per 32 bf16)
            asm volatile("s_wait_asynccnt 6" ::: "memory");   // my 6 loads for buf[cur] done
        } else {
            asm volatile("s_wait_asynccnt 0" ::: "memory");
        }
        __syncthreads();                       // buf[cur] ready for all waves

        // preload ALL fragments (distinct registers) so DS loads pipeline
        // ahead of the WMMA chain with partial dscnt waits
        Frag a[4], b[4];
#pragma unroll
        for (int mt = 0; mt < 4; ++mt) {
            int row = wm * 64 + mt * 16 + ml;
#pragma unroll
            for (int j = 0; j < 8; ++j) a[mt].u[j] = lA[cur][row * 16 + kcol(j, hf)];
        }
#pragma unroll
        for (int nt = 0; nt < 4; ++nt) {
            int row = wn * 64 + nt * 16 + ml;
#pragma unroll
            for (int j = 0; j < 8; ++j) b[nt].u[j] = lB[cur][row * 16 + kcol(j, hf)];
        }
#pragma unroll
        for (int nt = 0; nt < 4; ++nt)
#pragma unroll
            for (int mt = 0; mt < 4; ++mt)
                acc[mt][nt] = WMMA_BF16(a[mt].v, b[nt].v, acc[mt][nt]);
    }

    // Epilogue (C layout: row = i + 8*half, col = lane%16 within 16x16 tile)
#pragma unroll
    for (int mt = 0; mt < 4; ++mt) {
#pragma unroll
        for (int nt = 0; nt < 4; ++nt) {
#pragma unroll
            for (int i = 0; i < 8; ++i) {
                int row = m0 + wm * 64 + mt * 16 + hf * 8 + i;
                int col = n0 + wn * 64 + nt * 16 + ml;
                float val = acc[mt][nt][i];
                if constexpr (HAS_BIAS) val += bias[col];
                if constexpr (ACT_GELU) val = 0.5f * val * (1.0f + erff(val * 0.70710678118654752f));
                if constexpr (HAS_RESID) val += resid[(size_t)row * N + col];
                if constexpr (OUT_F) outF[(size_t)row * N + col] = val;
                if constexpr (OUT_B) outB[(size_t)row * N + col] = f2bf(val);
            }
        }
    }
}

// ---------------------------------------------------------------------------
// Flash attention: grid = (S/128, B*H), block = 256 threads (8 waves)
// Each wave owns 16 query rows (128 per block); Bc=64, HD=64, online softmax.
// ---------------------------------------------------------------------------
__global__ __launch_bounds__(256) void flash_attn(const unsigned short* __restrict__ qg,
                                                  const unsigned short* __restrict__ kg,
                                                  const unsigned short* __restrict__ vg,
                                                  const float* __restrict__ pad,
                                                  unsigned short* __restrict__ out) {
    __shared__ __attribute__((aligned(16))) unsigned int   lK[64 * 32];      // [j][d] row-major
    __shared__ __attribute__((aligned(16))) unsigned short lV[64 * 64];      // transposed [d][j]
    __shared__ __attribute__((aligned(16))) unsigned short lP[8][16 * 64];   // per-wave P staging

    const int tid  = threadIdx.x;
    const int wave = tid >> 5;
    const int lane = tid & 31;
    const int hf   = lane >> 4;
    const int ml   = lane & 15;
    const int qb   = blockIdx.x;
    const int bh   = blockIdx.y;
    const int b    = bh >> 4;
    const int h    = bh & 15;

    const unsigned int* q_u = (const unsigned int*)qg;
    const unsigned int* k_u = (const unsigned int*)kg;
    const unsigned int* v_u = (const unsigned int*)vg;
    const int Du = 512;   // D/2 uints per token row

    // Q fragments (A-layout), loaded once: rows = qb*128 + wave*16 + ml
    Frag qA[2];
    {
        const size_t qrow = (size_t)(b * 2048 + qb * 128 + wave * 16 + ml);
#pragma unroll
        for (int j = 0; j < 8; ++j) {
            int c = kcol(j, hf);
            qA[0].u[j] = q_u[qrow * Du + h * 32 + c];
            qA[1].u[j] = q_u[qrow * Du + h * 32 + 16 + c];
        }
    }

    float mrow[8], lrow[8];
    f32x8 oacc[4] = {};
#pragma unroll
    for (int i = 0; i < 8; ++i) { mrow[i] = -1e30f; lrow[i] = 0.f; }

    for (int jb = 0; jb < 32; ++jb) {
        __syncthreads();
        // Cooperative load of K tile (row-major) and V tile (transposed)
#pragma unroll
        for (int i = 0; i < 8; ++i) {
            int idx = i * 256 + tid;          // 0..2047
            int r = idx >> 5, c = idx & 31;
            size_t grow = (size_t)(b * 2048 + jb * 64 + r);
            lK[idx] = k_u[grow * Du + h * 32 + c];
            unsigned int vv = v_u[grow * Du + h * 32 + c];
            int d0 = c * 2;
            lV[d0 * 64 + r]       = (unsigned short)(vv & 0xffffu);
            lV[(d0 + 1) * 64 + r] = (unsigned short)(vv >> 16);
        }
        __syncthreads();

        // scores S = Q @ K^T  (16x64 per wave); preload all K fragments
        f32x8 sc[4] = {};
        {
            Frag kb[8];
#pragma unroll
            for (int nt = 0; nt < 4; ++nt) {
                int row = nt * 16 + ml;
#pragma unroll
                for (int j = 0; j < 8; ++j) {
                    int c = kcol(j, hf);
                    kb[nt * 2].u[j]     = lK[row * 32 + c];
                    kb[nt * 2 + 1].u[j] = lK[row * 32 + 16 + c];
                }
            }
#pragma unroll
            for (int nt = 0; nt < 4; ++nt) {
                sc[nt] = WMMA_BF16(qA[0].v, kb[nt * 2].v, sc[nt]);
                sc[nt] = WMMA_BF16(qA[1].v, kb[nt * 2 + 1].v, sc[nt]);
            }
        }

        float mval[4];
#pragma unroll
        for (int nt = 0; nt < 4; ++nt) {
            int col = jb * 64 + nt * 16 + ml;
            mval[nt] = (1.0f - pad[b * 2048 + col]) * (-10000.0f);
        }

        // online softmax; rows live at (i + 8*half), cols across 16 lanes
#pragma unroll
        for (int i = 0; i < 8; ++i) {
            float s[4];
            float rm = -1e30f;
#pragma unroll
            for (int nt = 0; nt < 4; ++nt) {
                s[nt] = sc[nt][i] * 0.125f + mval[nt];
                rm = fmaxf(rm, s[nt]);
            }
            for (int off = 1; off < 16; off <<= 1) rm = fmaxf(rm, __shfl_xor(rm, off, 16));
            float nm  = fmaxf(mrow[i], rm);
            float scl = __expf(mrow[i] - nm);
            mrow[i] = nm;
            float rs = 0.f;
            int prow = (hf * 8 + i) * 64;
#pragma unroll
            for (int nt = 0; nt < 4; ++nt) {
                float p = __expf(s[nt] - nm);
                rs += p;
                lP[wave][prow + nt * 16 + ml] = f2bf(p);
            }
            for (int off = 1; off < 16; off <<= 1) rs += __shfl_xor(rs, off, 16);
            lrow[i] = lrow[i] * scl + rs;
#pragma unroll
            for (int nt = 0; nt < 4; ++nt) oacc[nt][i] *= scl;
        }

        // per-wave LDS RAW: DS ops in-order within the wave, but be explicit
        asm volatile("s_wait_dscnt 0" ::: "memory");

        // reload P as A-fragments, V^T as B-fragments; O += P @ V
        Frag pA[2];
        const unsigned int* lPu = (const unsigned int*)&lP[wave][0];
#pragma unroll
        for (int j = 0; j < 8; ++j) {
            int c = kcol(j, hf);
            pA[0].u[j] = lPu[ml * 32 + c];
            pA[1].u[j] = lPu[ml * 32 + 16 + c];
        }
        const unsigned int* lVu = (const unsigned int*)lV;
        {
            Frag vb[8];
#pragma unroll
            for (int nt = 0; nt < 4; ++nt) {
                int dr = nt * 16 + ml;
#pragma unroll
                for (int j = 0; j < 8; ++j) {
                    int c = kcol(j, hf);
                    vb[nt * 2].u[j]     = lVu[dr * 32 + c];
                    vb[nt * 2 + 1].u[j] = lVu[dr * 32 + 16 + c];
                }
            }
#pragma unroll
            for (int nt = 0; nt < 4; ++nt) {
                oacc[nt] = WMMA_BF16(pA[0].v, vb[nt * 2].v, oacc[nt]);
                oacc[nt] = WMMA_BF16(pA[1].v, vb[nt * 2 + 1].v, oacc[nt]);
            }
        }
    }

    // write normalized output (bf16) to [B,S,D] with head offset
#pragma unroll
    for (int nt = 0; nt < 4; ++nt) {
#pragma unroll
        for (int i = 0; i < 8; ++i) {
            size_t grow = (size_t)(b * 2048 + qb * 128 + wave * 16 + hf * 8 + i);
            int col = h * 64 + nt * 16 + ml;
            out[grow * 1024 + col] = f2bf(oacc[nt][i] / lrow[i]);
        }
    }
}

// ---------------------------------------------------------------------------
// Host-side orchestration
// ---------------------------------------------------------------------------
extern "C" void kernel_launch(void* const* d_in, const int* in_sizes, int n_in,
                              void* d_out, int out_size, void* d_ws, size_t ws_size,
                              hipStream_t stream) {
    (void)in_sizes; (void)n_in; (void)out_size; (void)ws_size;
    constexpr int Bq = 4, S = 2048, D = 1024, H = 16, FF = 2048;
    constexpr int M = Bq * S;                       // 8192

    const float* x    = (const float*)d_in[0];
    const float* pad  = (const float*)d_in[1];
    const float* Wq   = (const float*)d_in[2];
    const float* Wk   = (const float*)d_in[3];
    const float* Wv   = (const float*)d_in[4];
    const float* Wu   = (const float*)d_in[5];
    const float* bu   = (const float*)d_in[6];
    const float* W1   = (const float*)d_in[7];
    const float* b1   = (const float*)d_in[8];
    const float* W2   = (const float*)d_in[9];
    const float* b2   = (const float*)d_in[10];
    const float* g1   = (const float*)d_in[11];
    const float* bt1  = (const float*)d_in[12];
    const float* g2   = (const float*)d_in[13];
    const float* bt2  = (const float*)d_in[14];
    float* out = (float*)d_out;

    // workspace layout (bytes)
    char* ws = (char*)d_ws;
    size_t off = 0;
    unsigned short* wq_b = (unsigned short*)(ws + off); off += (size_t)D * D * 2;    // 2MB
    unsigned short* wk_b = (unsigned short*)(ws + off); off += (size_t)D * D * 2;
    unsigned short* wv_b = (unsigned short*)(ws + off); off += (size_t)D * D * 2;
    unsigned short* wu_b = (unsigned short*)(ws + off); off += (size_t)D * D * 2;
    unsigned short* w1_b = (unsigned short*)(ws + off); off += (size_t)FF * D * 2;   // 4MB
    unsigned short* w2_b = (unsigned short*)(ws + off); off += (size_t)D * FF * 2;   // 4MB
    unsigned short* xn_b = (unsigned short*)(ws + off); off += (size_t)M * D * 2;    // 16MB (reused as xm)
    unsigned short* q_b  = (unsigned short*)(ws + off); off += (size_t)M * D * 2;    // 16MB (q_b..k_b reused as h)
    unsigned short* k_b  = (unsigned short*)(ws + off); off += (size_t)M * D * 2;
    unsigned short* v_b  = (unsigned short*)(ws + off); off += (size_t)M * D * 2;
    unsigned short* ao_b = (unsigned short*)(ws + off); off += (size_t)M * D * 2;
    float*          x2   = (float*)(ws + off);          off += (size_t)M * D * 4;    // 32MB
    unsigned short* xm_b = xn_b;          // xn dead after QKV
    unsigned short* h_b  = q_b;           // q+k slots (32MB contiguous) dead after attention

    // 1) weights -> bf16
    cvt_f32_bf16<<<2048, 256, 0, stream>>>(Wq, wq_b, D * D);
    cvt_f32_bf16<<<2048, 256, 0, stream>>>(Wk, wk_b, D * D);
    cvt_f32_bf16<<<2048, 256, 0, stream>>>(Wv, wv_b, D * D);
    cvt_f32_bf16<<<2048, 256, 0, stream>>>(Wu, wu_b, D * D);
    cvt_f32_bf16<<<2048, 256, 0, stream>>>(W1, w1_b, FF * D);
    cvt_f32_bf16<<<2048, 256, 0, stream>>>(W2, w2_b, D * FF);

    // 2) LN1
    ln_bf16<<<M, 256, 0, stream>>>(x, g1, bt1, xn_b);

    // 3) QKV projections: plain GEMM -> bf16
    dim3 gD(D / 128, M / 256);            // (8, 32)
    gemm_bf16<false, false, false, false, true><<<gD, 256, 0, stream>>>(
        xn_b, wq_b, nullptr, nullptr, nullptr, q_b, M, D, D);
    gemm_bf16<false, false, false, false, true><<<gD, 256, 0, stream>>>(
        xn_b, wk_b, nullptr, nullptr, nullptr, k_b, M, D, D);
    gemm_bf16<false, false, false, false, true><<<gD, 256, 0, stream>>>(
        xn_b, wv_b, nullptr, nullptr, nullptr, v_b, M, D, D);

    // 4) attention
    dim3 gA(S / 128, Bq * H);             // (16, 64)
    flash_attn<<<gA, 256, 0, stream>>>(q_b, k_b, v_b, pad, ao_b);

    // 5) output projection + bias + residual -> x2 (fp32)
    gemm_bf16<true, false, true, true, false><<<gD, 256, 0, stream>>>(
        ao_b, wu_b, bu, x, x2, nullptr, M, D, D);

    // 6) LN2
    ln_bf16<<<M, 256, 0, stream>>>(x2, g2, bt2, xm_b);

    // 7) FFN1: gelu(xm @ W1^T + b1) -> h (bf16)
    dim3 gF1(FF / 128, M / 256);          // (16, 32)
    gemm_bf16<true, true, false, false, true><<<gF1, 256, 0, stream>>>(
        xm_b, w1_b, b1, nullptr, nullptr, h_b, M, FF, D);

    // 8) FFN2: h @ W2^T + b2 + x2 -> out (fp32)
    gemm_bf16<true, false, true, true, false><<<gD, 256, 0, stream>>>(
        h_b, w2_b, b2, x2, out, nullptr, M, D, FF);
}
